// MultiHeadAttention_12369505813136
// MI455X (gfx1250) — matrix operand order repaired and checked by reading, not compiled, not run on hardware
//
#include <hip/hip_runtime.h>

// ---------------- problem constants (B=2, S=2048, D=1024, H=16, hd=64) ------
#define BB   2
#define SS   2048
#define DD   1024
#define HH   16
#define HD   64
#define MTOK (BB * SS)                 // 4096 tokens
// scores scaled by 1/sqrt(Sk) (reference!) and by log2(e) so softmax runs in
// exp2 domain; both folded into the Q projection output.
#define Q_SCALE (0.02209708691207961f * 1.4426950408889634f)

typedef __bf16 bf16;
typedef __attribute__((ext_vector_type(16))) __bf16 bf16x16;
typedef __attribute__((ext_vector_type(8)))  __bf16 bf16x8;
typedef __attribute__((ext_vector_type(8)))  float  f32x8;

#define DEV static __device__ __forceinline__

// ---- CDNA5 sync primitives (raw, no implicit vmem fences) ------------------
DEV void wg_barrier() {
  asm volatile("s_barrier_signal -1\n\ts_barrier_wait -1" ::: "memory");
}
DEV void async_wait1() { asm volatile("s_wait_asynccnt 0x1" ::: "memory"); }
DEV void ds_wait0()    { asm volatile("s_wait_dscnt 0x0"    ::: "memory"); }
DEV void compiler_fence() { asm volatile("" ::: "memory"); }

// ---- WMMA helpers -----------------------------------------------------------
DEV f32x8 wmma_bf16(bf16x16 a, bf16x16 b, f32x8 c) {
  return __builtin_amdgcn_wmma_f32_16x16x32_bf16(false, a, false, b, (short)0, c,
                                                 false, false);
}

// A-matrix 16x32 bf16 fragment (ISA 7.12.2): half-wave kg = (lane>=16)?8:0,
// elements 0..7 at p[0..7] (K=kg..), elements 8..15 at p[16..23] (K=16+kg..).
DEV bf16x16 load_a_frag(const bf16* p) {
  bf16x8 lo = *(const bf16x8*)(p);
  bf16x8 hi = *(const bf16x8*)(p + 16);
  bf16x16 r;
#pragma unroll
  for (int i = 0; i < 8; ++i) { r[i] = lo[i]; r[i + 8] = hi[i]; }
  return r;
}

// B-matrix 32x16 bf16 fragment: N = lane%16, kg = (lane>=16 ? 8 : 0); elements
// 0..7 at p[0..7] (K=kg..), 8..15 at p[16..23] (K=16+kg..); p = col_base + kb.
DEV bf16x16 load_b_frag(const bf16* p) {
  bf16x8 lo = *(const bf16x8*)(p);
  bf16x8 hi = *(const bf16x8*)(p + 16);
  bf16x16 r;
#pragma unroll
  for (int i = 0; i < 8; ++i) { r[i] = lo[i]; r[i + 8] = hi[i]; }
  return r;
}

// xor-lane max within 16-lane halves via ds_swizzle (group-of-32 xor mode:
// offset = and_mask(0x1f) | or_mask(0)<<5 | xor_mask<<10)
template <int XMASK>
DEV float swz_max(float v) {
  int o = __builtin_amdgcn_ds_swizzle(__float_as_int(v), 0x1f | (XMASK << 10));
  return fmaxf(v, __int_as_float(o));
}
DEV float halfmax(float v) {
  v = swz_max<1>(v); v = swz_max<2>(v); v = swz_max<4>(v); v = swz_max<8>(v);
  return v;
}

// ---------------- cast / transpose kernels ----------------------------------
__global__ void cast_f32_to_bf16(const float* __restrict__ src,
                                 bf16* __restrict__ dst, int n) {
  int i = blockIdx.x * blockDim.x + threadIdx.x;
  if (i < n) dst[i] = (bf16)src[i];
}

// W is KxN row-major; produce Wt = NxK row-major in bf16 (K-contiguous rows)
__global__ void transpose_cast_bf16(const float* __restrict__ src,
                                    bf16* __restrict__ dst, int K, int N) {
  int i = blockIdx.x * blockDim.x + threadIdx.x;
  if (i >= K * N) return;
  int n = i / K, k = i % K;
  dst[i] = (bf16)src[(size_t)k * N + n];
}

// ---------------- GEMM: C[MTOK,DD] = (A[MTOK,DD] @ Bt^T + bias) * outScale --
// block = 128 threads (4 waves); block tile 32(M) x 256(N); wave tile 32x64.
// A tile (32x32) staged to LDS by async DMA, 2 buffers alternated structurally
// (k-loop unrolled by 64); B fragments register double-buffered, no copies.
// MODE 0: bf16 row-major store | 1: bf16 -> Vt[b][h][d][s] | 2: f32 row-major
template <int MODE>
__global__ __launch_bounds__(128) void gemm_bf16_wmma(
    const bf16* __restrict__ A, const bf16* __restrict__ Bt,
    const float* __restrict__ bias, void* __restrict__ Cout, float outScale) {
  constexpr int N = DD, K = DD;
  __shared__ alignas(16) bf16 ldsA[2][32 * 32];  // 2 x 2KB

  const int tid  = threadIdx.x;
  const int wave = tid >> 5;
  const int lane = tid & 31;
  const int ph   = lane >> 4;
  const int l16  = lane & 15;
  const int mBase = blockIdx.y * 32;
  const int nBase = blockIdx.x * 256 + wave * 64;
  const int kg = ph * 8;
  const int kb = ph * 8;

  // --- async A-tile staging: 128 threads x 16B = 2KB = 32x32 bf16 -----------
  const int arow = tid >> 2;
  const int acol = (tid & 3) * 8;
  const unsigned ldsBase = (unsigned)(size_t)(&ldsA[0][0]);  // flat[31:0]==LDS off
  const bf16* aSrcRow = A + (size_t)(mBase + arow) * K + acol;

  auto issueA = [&](int bufSel, int k0) {
    unsigned ldst = ldsBase + (unsigned)(bufSel * 2048 + tid * 16);
    const bf16* g = aSrcRow + k0;
    asm volatile("global_load_async_to_lds_b128 %0, %1, off"
                 :: "v"(ldst), "v"(g) : "memory");
  };

  const bf16* bCol[4];
#pragma unroll
  for (int f = 0; f < 4; ++f)
    bCol[f] = Bt + (size_t)(nBase + f * 16 + l16) * K + kb;

  const f32x8 zero = {0.f, 0.f, 0.f, 0.f, 0.f, 0.f, 0.f, 0.f};
  f32x8 acc[2][4];
#pragma unroll
  for (int ms = 0; ms < 2; ++ms)
#pragma unroll
    for (int f = 0; f < 4; ++f) acc[ms][f] = zero;

  bf16x16 b0[4], b1[4];

  // prologue: stage k=0 -> buf0, k=32 -> buf1; B frags for both
  issueA(0, 0);
#pragma unroll
  for (int f = 0; f < 4; ++f) b0[f] = load_b_frag(bCol[f] + 0);
  issueA(1, 32);
#pragma unroll
  for (int f = 0; f < 4; ++f) b1[f] = load_b_frag(bCol[f] + 32);
  async_wait1();  // ASYNC completes in order: buf0 landed (mine)
  wg_barrier();   // everyone's buf0 landed

  for (int k0 = 0; k0 < K; k0 += 64) {
    // ---- stage 0: consume buf0 / b0 (k = k0) ------------------------------
    {
      bf16x16 af0 = load_a_frag(&ldsA[0][(l16)      * 32 + kg]);
      bf16x16 af1 = load_a_frag(&ldsA[0][(l16 + 16) * 32 + kg]);
#pragma unroll
      for (int f = 0; f < 4; ++f) {
        acc[0][f] = wmma_bf16(af0, b0[f], acc[0][f]);
        acc[1][f] = wmma_bf16(af1, b0[f], acc[1][f]);
      }
    }
    wg_barrier();                              // all waves done reading buf0
    const int kp0 = (k0 + 64 < K) ? k0 + 64 : 0;   // uniform clamp (s_cselect)
    issueA(0, kp0);
#pragma unroll
    for (int f = 0; f < 4; ++f) b0[f] = load_b_frag(bCol[f] + kp0);
    async_wait1();                             // buf1 (older async) landed
    wg_barrier();                              // everyone's buf1 landed

    // ---- stage 1: consume buf1 / b1 (k = k0+32) ---------------------------
    {
      bf16x16 af0 = load_a_frag(&ldsA[1][(l16)      * 32 + kg]);
      bf16x16 af1 = load_a_frag(&ldsA[1][(l16 + 16) * 32 + kg]);
#pragma unroll
      for (int f = 0; f < 4; ++f) {
        acc[0][f] = wmma_bf16(af0, b1[f], acc[0][f]);
        acc[1][f] = wmma_bf16(af1, b1[f], acc[1][f]);
      }
    }
    wg_barrier();                              // all waves done reading buf1
    const int kp1 = (k0 + 96 < K) ? k0 + 96 : 0;
    issueA(1, kp1);
#pragma unroll
    for (int f = 0; f < 4; ++f) b1[f] = load_b_frag(bCol[f] + kp1);
    async_wait1();                             // buf0 (k0+64) landed
    wg_barrier();
  }

#pragma unroll
  for (int f = 0; f < 4; ++f) {
    const int n = nBase + f * 16 + l16;
    const float bn = bias[n];
#pragma unroll
    for (int ms = 0; ms < 2; ++ms) {
#pragma unroll
      for (int r = 0; r < 8; ++r) {
        const int m = mBase + ms * 16 + r + ph * 8;  // lanes>=16 hold M=r+8
        const float v = (acc[ms][f][r] + bn) * outScale;
        if (MODE == 0) {
          ((bf16*)Cout)[(size_t)m * N + n] = (bf16)v;
        } else if (MODE == 1) {
          const int b = m / SS, s = m % SS;
          const int h = n / HD, d = n % HD;
          ((bf16*)Cout)[((size_t)((b * HH + h) * HD + d)) * SS + s] = (bf16)v;
        } else {
          ((float*)Cout)[(size_t)m * N + n] = v;
        }
      }
    }
  }
}

// ---------------- flash attention (causal, exp2-domain) ---------------------
// grid = (S/16, H, B), block = 32 (one wave -> 16 queries x full hd=64).
// Q arrives pre-scaled by 1/sqrt(Sk)*log2(e).  Key loop split: full chunks
// carry no mask code; exactly one edge chunk handles the causal diagonal.
__global__ __launch_bounds__(32) void flash_attn_wmma(
    const bf16* __restrict__ Qb,   // [MTOK][DD]  (pre-scaled)
    const bf16* __restrict__ Kb,   // [MTOK][DD]
    const bf16* __restrict__ Vt,   // [B][H][HD][SS]
    bf16* __restrict__ Ob) {       // [MTOK][DD]
  __shared__ alignas(16) bf16 ldsP[16 * 32];  // P tile C-layout -> A-layout

  const int lane = threadIdx.x & 31;
  const int ph   = lane >> 4;
  const int l16  = lane & 15;
  const int qBase = blockIdx.x * 16;
  const int h = blockIdx.y;
  const int b = blockIdx.z;
  const int kg = ph * 8;
  const int kb = ph * 8;

  const bf16* qRow = Qb + (size_t)(b * SS + qBase + l16) * DD + h * HD;
  const bf16x16 qf0 = load_a_frag(qRow + 0  + kg);
  const bf16x16 qf1 = load_a_frag(qRow + 32 + kg);

  const f32x8 zero = {0.f, 0.f, 0.f, 0.f, 0.f, 0.f, 0.f, 0.f};
  bf16x16 onesB;
#pragma unroll
  for (int i = 0; i < 16; ++i) onesB[i] = (bf16)1.0f;

  f32x8 o[4];
#pragma unroll
  for (int f = 0; f < 4; ++f) o[f] = zero;
  float mi[8], li[8];
#pragma unroll
  for (int r = 0; r < 8; ++r) { mi[r] = -__builtin_inff(); li[r] = 0.f; }

  auto kPtr = [&](int kc, int nf, int t) -> const bf16* {
    return Kb + (size_t)(b * SS + kc + nf * 16 + l16) * DD + h * HD + t * 32 + kb;
  };
  const bf16* vHead = Vt + (size_t)((b * HH + h) * HD) * SS;

  bf16x16 kcur[2][2], knext[2][2];
#pragma unroll
  for (int nf = 0; nf < 2; ++nf)
#pragma unroll
    for (int t = 0; t < 2; ++t) kcur[nf][t] = load_b_frag(kPtr(0, nf, t));

  // one 32-key chunk; `masked` is a compile-time constant at each call site
  auto chunk = [&](int kc, int kcn, bool masked) __attribute__((always_inline)) {
    // ---- scores: 16 queries x 32 keys (already exp2-domain scaled) --------
    f32x8 sf[2];
    sf[0] = zero; sf[1] = zero;
#pragma unroll
    for (int nf = 0; nf < 2; ++nf) {
      sf[nf] = wmma_bf16(qf0, kcur[nf][0], sf[nf]);
      sf[nf] = wmma_bf16(qf1, kcur[nf][1], sf[nf]);
    }

    // ---- prefetch: this chunk's V frags + next chunk's K frags ------------
    bf16x16 vf[4];
#pragma unroll
    for (int f = 0; f < 4; ++f)
      vf[f] = load_b_frag(vHead + (size_t)(f * 16 + l16) * SS + kc + kb);
#pragma unroll
    for (int nf = 0; nf < 2; ++nf)
#pragma unroll
      for (int t = 0; t < 2; ++t) knext[nf][t] = load_b_frag(kPtr(kcn, nf, t));

    if (masked) {
#pragma unroll
      for (int nf = 0; nf < 2; ++nf) {
        const int k = kc + nf * 16 + l16;
#pragma unroll
        for (int r = 0; r < 8; ++r) {
          const int q = qBase + r + ph * 8;
          if (k > q) sf[nf][r] = -__builtin_inff();
        }
      }
    }

    // ---- online softmax (exp2 domain) -------------------------------------
    float mnew[8], alpha[8];
#pragma unroll
    for (int r = 0; r < 8; ++r) {
      float v = halfmax(fmaxf(sf[0][r], sf[1][r]));
      mnew[r]  = fmaxf(mi[r], v);
      alpha[r] = __builtin_amdgcn_exp2f(mi[r] - mnew[r]);  // first chunk: 0
      mi[r] = mnew[r];
    }
#pragma unroll
    for (int nf = 0; nf < 2; ++nf)
#pragma unroll
      for (int r = 0; r < 8; ++r)
        sf[nf][r] = __builtin_amdgcn_exp2f(sf[nf][r] - mnew[r]);
#pragma unroll
    for (int f = 0; f < 4; ++f)
#pragma unroll
      for (int r = 0; r < 8; ++r) o[f][r] *= alpha[r];

    // ---- relayout P (C layout) -> A fragment via LDS ----------------------
    // single-wave block: DS in-order per wave; dscnt drain + compiler fences.
    compiler_fence();
#pragma unroll
    for (int nf = 0; nf < 2; ++nf)
#pragma unroll
      for (int r = 0; r < 8; ++r)
        ldsP[(r + ph * 8) * 32 + nf * 16 + l16] = (bf16)sf[nf][r];
    ds_wait0();
    const bf16x16 pf = load_a_frag(&ldsP[l16 * 32 + kg]);
    compiler_fence();

    // ---- row sums via WMMA: D[m][n] = sum_k P[m][k]*1, same slot layout ---
    f32x8 rs = wmma_bf16(pf, onesB, zero);
#pragma unroll
    for (int r = 0; r < 8; ++r) li[r] = li[r] * alpha[r] + rs[r];

    // ---- O += P @ V (V frags already in registers) ------------------------
#pragma unroll
    for (int f = 0; f < 4; ++f) o[f] = wmma_bf16(pf, vf[f], o[f]);

#pragma unroll
    for (int nf = 0; nf < 2; ++nf)
#pragma unroll
      for (int t = 0; t < 2; ++t) kcur[nf][t] = knext[nf][t];
  };

  // full (unmasked) chunks, then exactly one causal edge chunk
  const int kFull = qBase & ~31;
  for (int kc = 0; kc < kFull; kc += 32) chunk(kc, kc + 32, false);
  chunk(kFull, 0, true);

  // ---- normalize + store O as [b,s,h,hd] bf16 -----------------------------
  float invl[8];
#pragma unroll
  for (int r = 0; r < 8; ++r) invl[r] = 1.f / li[r];
#pragma unroll
  for (int f = 0; f < 4; ++f)
#pragma unroll
    for (int r = 0; r < 8; ++r) {
      const float v = o[f][r] * invl[r];
      const int m = b * SS + qBase + r + ph * 8;
      Ob[(size_t)m * DD + h * HD + f * 16 + l16] = (bf16)v;
    }
}

// ---------------- host driver ------------------------------------------------
extern "C" void kernel_launch(void* const* d_in, const int* in_sizes, int n_in,
                              void* d_out, int out_size, void* d_ws,
                              size_t ws_size, hipStream_t stream) {
  const float* query = (const float*)d_in[0];
  const float* key   = (const float*)d_in[1];
  const float* value = (const float*)d_in[2];
  const float* Wq = (const float*)d_in[3];  const float* bq = (const float*)d_in[4];
  const float* Wk = (const float*)d_in[5];  const float* bk = (const float*)d_in[6];
  const float* Wv = (const float*)d_in[7];  const float* bv = (const float*)d_in[8];
  const float* Wo = (const float*)d_in[9];  const float* bo = (const float*)d_in[10];

  char* ws = (char*)d_ws;
  size_t off = 0;
  auto alloc = [&](size_t bytes) -> void* {
    void* p = ws + off;
    off += (bytes + 255) & ~(size_t)255;
    return p;
  };
  const size_t tokB = (size_t)MTOK * DD * sizeof(bf16);  // 8 MB
  const size_t wB   = (size_t)DD * DD * sizeof(bf16);    // 2 MB

  bf16* Xq  = (bf16*)alloc(tokB);
  bf16* Xk  = (bf16*)alloc(tokB);
  bf16* Xv  = (bf16*)alloc(tokB);
  bf16* Wqt = (bf16*)alloc(wB);
  bf16* Wkt = (bf16*)alloc(wB);
  bf16* Wvt = (bf16*)alloc(wB);
  bf16* Wot = (bf16*)alloc(wB);
  bf16* Qb  = (bf16*)alloc(tokB);
  bf16* Kbf = (bf16*)alloc(tokB);
  bf16* Vtb = (bf16*)alloc(tokB);
  bf16* Ob  = Xq;  // Xq is dead after the Q projection; reuse for attention out

  const int nTok = MTOK * DD;      // 4,194,304
  const int nW   = DD * DD;        // 1,048,576

  cast_f32_to_bf16<<<(nTok + 255) / 256, 256, 0, stream>>>(query, Xq, nTok);
  cast_f32_to_bf16<<<(nTok + 255) / 256, 256, 0, stream>>>(key,   Xk, nTok);
  cast_f32_to_bf16<<<(nTok + 255) / 256, 256, 0, stream>>>(value, Xv, nTok);
  transpose_cast_bf16<<<(nW + 255) / 256, 256, 0, stream>>>(Wq, Wqt, DD, DD);
  transpose_cast_bf16<<<(nW + 255) / 256, 256, 0, stream>>>(Wk, Wkt, DD, DD);
  transpose_cast_bf16<<<(nW + 255) / 256, 256, 0, stream>>>(Wv, Wvt, DD, DD);
  transpose_cast_bf16<<<(nW + 255) / 256, 256, 0, stream>>>(Wo, Wot, DD, DD);

  dim3 ggrid(DD / 256, MTOK / 32);  // (4, 128)
  dim3 gblk(128);                   // 4 waves
  gemm_bf16_wmma<0><<<ggrid, gblk, 0, stream>>>(Xq, Wqt, bq, Qb,  Q_SCALE);
  gemm_bf16_wmma<0><<<ggrid, gblk, 0, stream>>>(Xk, Wkt, bk, Kbf, 1.0f);
  gemm_bf16_wmma<1><<<ggrid, gblk, 0, stream>>>(Xv, Wvt, bv, Vtb, 1.0f);

  dim3 fgrid(SS / 16, HH, BB);      // (128, 16, 2)
  flash_attn_wmma<<<fgrid, dim3(32), 0, stream>>>(Qb, Kbf, Vtb, Ob);

  gemm_bf16_wmma<2><<<ggrid, gblk, 0, stream>>>(Ob, Wot, bo, d_out, 1.0f);
}
